// Attention_85925115724482
// MI455X (gfx1250) — matrix operand order repaired and checked by reading, not compiled, not run on hardware
//
#include <hip/hip_runtime.h>
#include <hip/hip_bf16.h>

typedef __attribute__((ext_vector_type(16))) _Float16 v16h;
typedef __attribute__((ext_vector_type(8)))  _Float16 v8h;
typedef __attribute__((ext_vector_type(8)))  float    v8f;

// Build a 16-half WMMA fragment from two 16-byte chunks (LDS or global).
__device__ __forceinline__ v16h frag16(const _Float16* p0, const _Float16* p1) {
    union { v16h v; v8h h[2]; } u;
    u.h[0] = *(const v8h*)p0;
    u.h[1] = *(const v8h*)p1;
    return u.v;
}

// ---------------------------------------------------------------------------
// GEMM: out[m, p] = sum_c W[m,c] * X[c,p] + bias[m], per batch (blockIdx.z).
//   X layout: (batch, KD, 4096) with pixel index contiguous.
//   EPI==0: scatter f16 to ws layout ((b*8+h)*4096 + p)*32 + d   (m = h*32+d)
//   EPI==1: f32 to d_out layout (b, M, 4096)
// Workgroup tile 64(M) x 128(N); 8 waves as 2x4; wave tile 32x32 = 2x2 WMMA.
// ---------------------------------------------------------------------------
template <typename T, int EPI, int KD>
__global__ __launch_bounds__(256)
void gemm_wmma(const T* __restrict__ X, const float* __restrict__ W,
               const float* __restrict__ bias, void* __restrict__ outp, int M) {
    constexpr int HW = 4096;
    __shared__ __align__(16) _Float16 Alds[64 * 32];    // [m][k]
    __shared__ __align__(16) _Float16 Blds[128 * 32];   // [n][k] (transposed)

    const int b  = blockIdx.z;
    const int m0 = blockIdx.x * 64;
    const int n0 = blockIdx.y * 128;
    const T* Xb  = X + (size_t)b * KD * HW;

    const int t    = threadIdx.x;
    const int lane = t & 31;
    const int w    = t >> 5;
    const int wm   = w >> 2;        // 0..1
    const int wn   = w & 3;         // 0..3
    const int hi   = (lane >> 4);   // lane half selects K sub-range
    const int lmod = lane & 15;

    v8f acc[2][2] = {};

#pragma unroll
    for (int k0 = 0; k0 < KD; k0 += 32) {
        // --- stage A tile (64x32) : W row-major, convert f32->f16 ---
        {
            const int i  = t >> 2;           // 0..63
            const int j0 = (t & 3) * 8;
            const float* src = W + (size_t)(m0 + i) * KD + k0 + j0;
            _Float16* dst = Alds + i * 32 + j0;
#pragma unroll
            for (int e = 0; e < 8; ++e) dst[e] = (_Float16)src[e];
        }
        // --- stage B tile (32x128) transposed into [n][k] ---
        {
            const int k   = t >> 3;          // 0..31
            const int nn0 = (t & 7) * 16;
            const T* src = Xb + (size_t)(k0 + k) * HW + n0 + nn0;
#pragma unroll
            for (int e = 0; e < 16; ++e) Blds[(nn0 + e) * 32 + k] = (_Float16)src[e];
        }
        __syncthreads();

        // A 16x32 f16 layout: lane = M; halves 0-7 -> K = hi*8 + 0..7,
        //                     halves 8-15 -> K = 16 + hi*8 + 0..7.
        v16h afrag[2], bfrag[2];
#pragma unroll
        for (int mt = 0; mt < 2; ++mt) {
            const _Float16* base = Alds + (wm * 32 + mt * 16 + lmod) * 32 + hi * 8;
            afrag[mt] = frag16(base, base + 16);
        }
        // B 32x16 f16 layout: lane = N; halves 0-7 -> K = hi*8 + 0..7, halves 8-15 -> K = 16 + hi*8 + 0..7.
#pragma unroll
        for (int nt = 0; nt < 2; ++nt) {
            const _Float16* base = Blds + (wn * 32 + nt * 16 + lmod) * 32 + hi * 8;
            bfrag[nt] = frag16(base, base + 16);
        }
#pragma unroll
        for (int mt = 0; mt < 2; ++mt)
#pragma unroll
            for (int nt = 0; nt < 2; ++nt)
                acc[mt][nt] = __builtin_amdgcn_wmma_f32_16x16x32_f16(
                    false, afrag[mt], false, bfrag[nt],
                    (short)0, acc[mt][nt], false, false);
        __syncthreads();
    }

    // C/D layout: VGPR r -> M = hi*8 + r, lane -> N = lmod.
#pragma unroll
    for (int mt = 0; mt < 2; ++mt)
#pragma unroll
        for (int nt = 0; nt < 2; ++nt)
#pragma unroll
            for (int r = 0; r < 8; ++r) {
                const int m = m0 + wm * 32 + mt * 16 + hi * 8 + r;
                const int n = n0 + wn * 32 + nt * 16 + lmod;
                const float val = acc[mt][nt][r] + bias[m];
                if (EPI == 0) {
                    _Float16* o = (_Float16*)outp;
                    const int h = m >> 5, d = m & 31;
                    o[((size_t)(b * 8 + h) * HW + n) * 32 + d] = (_Float16)val;
                } else {
                    float* o = (float*)outp;
                    o[((size_t)b * M + m) * HW + n] = val;
                }
            }
}

// ---------------------------------------------------------------------------
// Flash-style masked dense neighborhood attention (7x7 window, zero-padded
// borders kept in the softmax, matching the reference's zero-padded unfold).
//
// Workgroup = 8x16 query tile; wave w owns query row w (M-tile = 16 queries of
// one image row => their window union is halo rows [w, w+6] only).
// Columns for a wave: 7 halo rows * 24 (padded stride) = 168, padded to 192
// (6 chunks of 32). S kept in registers; two-pass softmax (no online rescale).
// ---------------------------------------------------------------------------
__global__ __launch_bounds__(256)
void attn_flash(const _Float16* __restrict__ Q, const _Float16* __restrict__ Kt,
                const _Float16* __restrict__ V, _Float16* __restrict__ attn_out) {
    constexpr int HW = 4096, D = 32;
    constexpr int TY = 8, TX = 16;        // queries per tile
    constexpr int HROWS = 14;             // halo rows (TY + 6)
    constexpr int HSTR = 24;              // padded halo row stride (22 valid)
    constexpr int PSZ = 368;              // halo pixel space (max read 359)
    constexpr int NC = 6;                 // 6 chunks of 32 columns (=192)
    __shared__ __align__(16) _Float16 Kl[PSZ * D];    // [pixel][d]
    __shared__ __align__(16) _Float16 VlT[D * PSZ];   // [d][pixel]
    __shared__ __align__(16) _Float16 Pl[8 * 16 * 32];// per-wave P chunk

    const int tile = blockIdx.x;          // 0..31 : 8(y) x 4(x) tiles
    const int head = blockIdx.y;
    const int b    = blockIdx.z;
    const int ty0  = (tile >> 2) * TY;
    const int tx0  = (tile & 3) * TX;
    const size_t baseHW = (size_t)(b * 8 + head) * HW;

    const int t    = threadIdx.x;
    const int lane = t & 31;
    const int w    = t >> 5;              // wave id == query row in tile
    const int hi   = lane >> 4;
    const int lmod = lane & 15;

    // ---- stage halo K (pixel-major) and V (transposed), zero-padded ----
    for (int p = t; p < PSZ; p += 256) {
        const int hy = p / HSTR, hx = p - hy * HSTR;
        const int y = ty0 - 3 + hy, x = tx0 - 3 + hx;
        const bool ok = (hy < HROWS) && (hx < 22) &&
                        ((unsigned)y < 64u) && ((unsigned)x < 64u);
        union { uint4 u[4]; _Float16 f[32]; } kv, vv;
        if (ok) {
            const uint4* ks = (const uint4*)(Kt + (baseHW + y * 64 + x) * D);
            const uint4* vs = (const uint4*)(V  + (baseHW + y * 64 + x) * D);
#pragma unroll
            for (int e = 0; e < 4; ++e) { kv.u[e] = ks[e]; vv.u[e] = vs[e]; }
        } else {
            const uint4 z = {0u, 0u, 0u, 0u};
#pragma unroll
            for (int e = 0; e < 4; ++e) { kv.u[e] = z; vv.u[e] = z; }
        }
        uint4* kd = (uint4*)(Kl + p * D);
#pragma unroll
        for (int e = 0; e < 4; ++e) kd[e] = kv.u[e];
#pragma unroll
        for (int d = 0; d < 32; ++d) VlT[d * PSZ + p] = vv.f[d];
    }
    __syncthreads();

    // ---- Q fragment for this wave's row, pre-scaled by 1/sqrt(dk) ----
    const int qy = ty0 + w;
    const _Float16* qg = Q + (baseHW + (size_t)qy * 64 + tx0 + lmod) * D + hi * 8;
    v16h qf = frag16(qg, qg + 16);
#pragma unroll
    for (int i = 0; i < 16; ++i) qf[i] = qf[i] * (_Float16)0.17677669529663687f;

    // ---- pass A: S = Q * K^T over 192 columns, kept in registers ----
    const v8f zz = {};
    v8f s[NC][2];
#pragma unroll
    for (int c = 0; c < NC; ++c)
#pragma unroll
        for (int st = 0; st < 2; ++st) {
            const _Float16* kb =
                Kl + (size_t)(w * HSTR + c * 32 + st * 16 + lmod) * D + hi * 8;
            v16h kf = frag16(kb, kb + 16);
            s[c][st] = __builtin_amdgcn_wmma_f32_16x16x32_f16(
                false, qf, false, kf, (short)0, zz, false, false);
        }

    // ---- window mask + row max (rows owned per C layout: M = hi*8 + r) ----
    float mx[8];
#pragma unroll
    for (int r = 0; r < 8; ++r) mx[r] = -1e30f;
#pragma unroll
    for (int c = 0; c < NC; ++c)
#pragma unroll
        for (int st = 0; st < 2; ++st) {
            const int cq = c * 32 + st * 16 + lmod;       // column id 0..191
            const int hy = cq / HSTR, hx = cq - hy * HSTR;
            const bool colok = (cq < 7 * HSTR);           // hy in [0,6]
#pragma unroll
            for (int r = 0; r < 8; ++r) {
                const int lxq = hi * 8 + r;               // query x in tile
                const bool valid = colok && ((unsigned)(hx - lxq) <= 6u);
                const float sv = valid ? s[c][st][r] : -1e30f;
                s[c][st][r] = sv;
                mx[r] = fmaxf(mx[r], sv);
            }
        }
#pragma unroll
    for (int r = 0; r < 8; ++r)
#pragma unroll
        for (int off = 8; off >= 1; off >>= 1)
            mx[r] = fmaxf(mx[r], __shfl_xor(mx[r], off, 32));

    // ---- pass B: P = exp(S - m) -> LDS -> WMMA against V^T ----
    float lsum[8];
#pragma unroll
    for (int r = 0; r < 8; ++r) lsum[r] = 0.f;
    v8f accO[2] = {zz, zz};
    _Float16* pw = Pl + w * 512;                          // [16 rows][32 cols]

#pragma unroll
    for (int c = 0; c < NC; ++c) {
#pragma unroll
        for (int st = 0; st < 2; ++st)
#pragma unroll
            for (int r = 0; r < 8; ++r) {
                const float p = __expf(s[c][st][r] - mx[r]);
                lsum[r] += p;
                pw[(hi * 8 + r) * 32 + st * 16 + lmod] = (_Float16)p;
            }
        const _Float16* pb = pw + lmod * 32 + hi * 8;
        v16h pf = frag16(pb, pb + 16);                    // A-frag: lane = query
#pragma unroll
        for (int dvt = 0; dvt < 2; ++dvt) {
            const _Float16* vb =
                VlT + (size_t)(dvt * 16 + lmod) * PSZ + w * HSTR + c * 32 + hi * 8;
            v16h vf = frag16(vb, vb + 16);                // B-frag: lane = dv
            accO[dvt] = __builtin_amdgcn_wmma_f32_16x16x32_f16(
                false, pf, false, vf, (short)0, accO[dvt], false, false);
        }
    }

#pragma unroll
    for (int r = 0; r < 8; ++r)
#pragma unroll
        for (int off = 8; off >= 1; off >>= 1)
            lsum[r] += __shfl_xor(lsum[r], off, 32);
    float inv[8];
#pragma unroll
    for (int r = 0; r < 8; ++r) inv[r] = 1.f / lsum[r];

    // ---- write attn ws (b, c = head*32 + d, p) for the O-projection GEMM ----
#pragma unroll
    for (int dvt = 0; dvt < 2; ++dvt)
#pragma unroll
        for (int r = 0; r < 8; ++r) {
            const int d = dvt * 16 + lmod;
            const int x = tx0 + hi * 8 + r;
            attn_out[((size_t)(b * 256 + head * 32 + d)) * HW + qy * 64 + x] =
                (_Float16)(accO[dvt][r] * inv[r]);
        }
}

// ---------------------------------------------------------------------------
extern "C" void kernel_launch(void* const* d_in, const int* in_sizes, int n_in,
                              void* d_out, int out_size, void* d_ws, size_t ws_size,
                              hipStream_t stream) {
    const float* x  = (const float*)d_in[0];
    const float* Wk = (const float*)d_in[1];
    const float* bk = (const float*)d_in[2];
    const float* Wq = (const float*)d_in[3];
    const float* bq = (const float*)d_in[4];
    const float* Wv = (const float*)d_in[5];
    const float* bv = (const float*)d_in[6];
    const float* Wo = (const float*)d_in[7];
    const float* bo = (const float*)d_in[8];

    const size_t QKV_ELEMS = (size_t)2 * 8 * 4096 * 32;   // 4 MB each (f16)
    _Float16* ws_q = (_Float16*)d_ws;
    _Float16* ws_k = ws_q + QKV_ELEMS;
    _Float16* ws_v = ws_k + QKV_ELEMS;
    _Float16* ws_a = ws_v + QKV_ELEMS;                    // (2, 256, 4096) f16

    const dim3 blk(256);
    const dim3 g_qkv(256 / 64, 4096 / 128, 2);
    gemm_wmma<float, 0, 128><<<g_qkv, blk, 0, stream>>>(x, Wq, bq, ws_q, 256);
    gemm_wmma<float, 0, 128><<<g_qkv, blk, 0, stream>>>(x, Wk, bk, ws_k, 256);
    gemm_wmma<float, 0, 128><<<g_qkv, blk, 0, stream>>>(x, Wv, bv, ws_v, 256);

    const dim3 g_at(32, 8, 2);
    attn_flash<<<g_at, blk, 0, stream>>>(ws_q, ws_k, ws_v, ws_a);

    const dim3 g_o(128 / 64, 4096 / 128, 2);
    gemm_wmma<_Float16, 1, 256><<<g_o, blk, 0, stream>>>(ws_a, Wo, bo, d_out, 128);
}
